// CausalSelfAttention_21509196219266
// MI455X (gfx1250) — matrix operand order repaired and checked by reading, not compiled, not run on hardware
//
#include <hip/hip_runtime.h>
#include <cstdint>

// ---------------------------------------------------------------------------
// CDNA5 (gfx1250) causal self-attention: bf16 WMMA pipeline, f32 accumulate.
// Register-blocked GEMMs (64x64/wave) + 32-row flash attention with
// async global->LDS staging of V (ASYNCcnt path).
// ---------------------------------------------------------------------------

#define DEVINL __device__ __forceinline__

typedef __attribute__((ext_vector_type(16))) __bf16 v16bf;
typedef __attribute__((ext_vector_type(8)))  float  v8f;

static constexpr int Bc = 2;
static constexpr int Tc = 2048;
static constexpr int Cc = 1024;
static constexpr int Hc = 16;
static constexpr int Dc = 64;
static constexpr int Mrows = Bc * Tc;     // 4096
static constexpr int Nqkv  = 3 * Cc;      // 3072

#define NEG_INF (-__builtin_inff())

DEVINL v8f zero8() {
  v8f z;
#pragma unroll
  for (int i = 0; i < 8; ++i) z[i] = 0.0f;
  return z;
}

DEVINL float rmax16(float v) {
#pragma unroll
  for (int m = 1; m < 16; m <<= 1) v = fmaxf(v, __shfl_xor(v, m, 32));
  return v;
}

DEVINL float rsum16(float v) {
#pragma unroll
  for (int m = 1; m < 16; m <<= 1) v += __shfl_xor(v, m, 32);
  return v;
}

// Build a v16bf fragment from two runs of 8 contiguous f32 (convert on load).
DEVINL v16bf frag_from_f32(const float* __restrict__ p0,
                           const float* __restrict__ p1) {
  union { float4 q[4]; float f[16]; } t;
  t.q[0] = ((const float4*)p0)[0];
  t.q[1] = ((const float4*)p0)[1];
  t.q[2] = ((const float4*)p1)[0];
  t.q[3] = ((const float4*)p1)[1];
  v16bf a;
#pragma unroll
  for (int i = 0; i < 16; ++i) a[i] = (__bf16)t.f[i];
  return a;
}

// Build a v16bf fragment from two runs of 8 contiguous bf16 (16B each).
DEVINL v16bf frag_from_bf16(const __bf16* __restrict__ p0,
                            const __bf16* __restrict__ p1) {
  union { v16bf v; uint4 u[2]; } a;
  a.u[0] = *(const uint4*)p0;
  a.u[1] = *(const uint4*)p1;
  return a.v;
}

DEVINL v8f wmma_bf16(v16bf a, v16bf b, v8f c) {
  // D = A(16x32 bf16) x B(32x16 bf16) + C(16x16 f32)
  return __builtin_amdgcn_wmma_f32_16x16x32_bf16(
      /*neg_a=*/false, a, /*neg_b=*/false, b,
      /*c_mod=*/(short)0, c, /*reuse_a=*/false, /*reuse_b=*/false);
}

// ---------------------------------------------------------------------------
// Kernel 1: QKV = X @ W_attn^T + b_attn  (f32 in, bf16 out)
// Each wave computes a 64x64 output block (4x4 WMMA tiles, 16 wmma / k-step).
// 4 waves per block tile 256 output columns.
// ---------------------------------------------------------------------------
__global__ __launch_bounds__(128) void qkv_gemm(
    const float* __restrict__ x, const float* __restrict__ Wa,
    const float* __restrict__ ba, __bf16* __restrict__ QKV) {
  const int lane = threadIdx.x & 31, wid = threadIdx.x >> 5;
  const int half = lane >> 4, l16 = lane & 15;
  const int tM64 = blockIdx.x / (Nqkv / 256);
  const int nW   = (blockIdx.x % (Nqkv / 256)) * 256 + wid * 64;
  const int rowBase = tM64 * 64;

  const float* xr[4];
  const float* wr[4];
#pragma unroll
  for (int mt = 0; mt < 4; ++mt)
    xr[mt] = x + (size_t)(rowBase + mt * 16 + l16) * Cc;
#pragma unroll
  for (int nt = 0; nt < 4; ++nt)
    wr[nt] = Wa + (size_t)(nW + nt * 16 + l16) * Cc;

  v8f acc[4][4];
#pragma unroll
  for (int mt = 0; mt < 4; ++mt)
#pragma unroll
    for (int nt = 0; nt < 4; ++nt) acc[mt][nt] = zero8();

  for (int kb = 0; kb < Cc; kb += 32) {
    v16bf a[4], b[4];
#pragma unroll
    for (int mt = 0; mt < 4; ++mt)
      a[mt] = frag_from_f32(xr[mt] + kb + half * 8,
                            xr[mt] + kb + 16 + half * 8);
#pragma unroll
    for (int nt = 0; nt < 4; ++nt)
      b[nt] = frag_from_f32(wr[nt] + kb + half * 8,
                            wr[nt] + kb + 16 + half * 8);
#pragma unroll
    for (int mt = 0; mt < 4; ++mt)
#pragma unroll
      for (int nt = 0; nt < 4; ++nt)
        acc[mt][nt] = wmma_bf16(a[mt], b[nt], acc[mt][nt]);
  }

#pragma unroll
  for (int nt = 0; nt < 4; ++nt) {
    const int n = nW + nt * 16 + l16;
    const float bias = ba[n];
#pragma unroll
    for (int mt = 0; mt < 4; ++mt) {
#pragma unroll
      for (int r = 0; r < 8; ++r) {
        const int row = rowBase + mt * 16 + r + 8 * half;  // D layout: M=r+8*half
        QKV[(size_t)row * Nqkv + n] = (__bf16)(acc[mt][nt][r] + bias);
      }
    }
  }
}

// ---------------------------------------------------------------------------
// Kernel 2: interleaved-pair RoPE on Q,K + repack to [B,H,T,D] bf16; copy V.
// One (b,h,t) per 64-thread block.
// ---------------------------------------------------------------------------
__global__ __launch_bounds__(64) void rope_reshape(
    const __bf16* __restrict__ QKV, __bf16* __restrict__ Qh,
    __bf16* __restrict__ Kh, __bf16* __restrict__ Vh) {
  const int blk = blockIdx.x;
  const int t  = blk % Tc;
  const int bh = blk / Tc;
  const int h  = bh % Hc;
  const int b  = bh / Hc;
  const int tid = threadIdx.x;

  const size_t row  = (size_t)(b * Tc + t) * Nqkv;
  const size_t orow = (size_t)(((size_t)b * Hc + h) * Tc + t) * Dc;
  const int cbase = h * Dc;

  if (tid < 32) {
    const int d0 = tid * 2;
    const float ang = (float)t * __powf(10000.0f, -((float)d0) / (float)Dc);
    const float cs = __cosf(ang), sn = __sinf(ang);

    float q0 = (float)QKV[row + cbase + d0];
    float q1 = (float)QKV[row + cbase + d0 + 1];
    Qh[orow + d0]     = (__bf16)(q0 * cs - q1 * sn);
    Qh[orow + d0 + 1] = (__bf16)(q1 * cs + q0 * sn);

    float k0 = (float)QKV[row + Cc + cbase + d0];
    float k1 = (float)QKV[row + Cc + cbase + d0 + 1];
    Kh[orow + d0]     = (__bf16)(k0 * cs - k1 * sn);
    Kh[orow + d0 + 1] = (__bf16)(k1 * cs + k0 * sn);
  }
  Vh[orow + tid] = QKV[row + 2 * Cc + cbase + tid];
}

// ---------------------------------------------------------------------------
// Kernel 3: flash attention (online softmax), causal, scale = 1/sqrt(64).
// 32 query rows per wave; 32-wide key blocks; 16 WMMA per key block.
// V block staged global->LDS with CDNA5 async LDS loads (ASYNCcnt).
// ---------------------------------------------------------------------------
__global__ __launch_bounds__(128) void flash_attn(
    const __bf16* __restrict__ Qh, const __bf16* __restrict__ Kh,
    const __bf16* __restrict__ Vh, __bf16* __restrict__ Yb) {
  __shared__ __align__(16) __bf16 Vs[4][32][64];  // per-wave V block (row-major)
  __shared__ __align__(16) __bf16 Ps[4][32][32];  // per-wave P tiles (bf16)

  const int lane = threadIdx.x & 31, wid = threadIdx.x >> 5;
  const int half = lane >> 4, l16 = lane & 15;
  const int t128 = blockIdx.x % (Tc / 128);
  const int bh   = blockIdx.x / (Tc / 128);
  const int h = bh % Hc, b = bh / Hc;
  const int qrow = (t128 * 4 + wid) * 32;
  const size_t base = (size_t)bh * Tc * Dc;

  // Q fragments (A layout): 2 row-tiles x 2 k-splits, loaded once.
  v16bf qa[2][2];
#pragma unroll
  for (int mt = 0; mt < 2; ++mt) {
    const __bf16* qp = Qh + base + (size_t)(qrow + mt * 16 + l16) * Dc;
    qa[mt][0] = frag_from_bf16(qp + half * 8,      qp + 16 + half * 8);
    qa[mt][1] = frag_from_bf16(qp + 32 + half * 8, qp + 48 + half * 8);
  }

  float m[2][8], l[2][8], alpha[2][8];
  v8f O[2][4];
#pragma unroll
  for (int mt = 0; mt < 2; ++mt) {
#pragma unroll
    for (int r = 0; r < 8; ++r) { m[mt][r] = NEG_INF; l[mt][r] = 0.0f; }
#pragma unroll
    for (int dt = 0; dt < 4; ++dt) O[mt][dt] = zero8();
  }

  const int nkb = (qrow + 31) / 32 + 1;   // wave-uniform causal trip count
  for (int kb = 0; kb < nkb; ++kb) {
    const int k0 = kb * 32;

    // Kick off async V staging (32 keys x 64 dims, 4KB) early: lane owns row
    // `lane`; same INST_OFFSET applies to both LDS and global addresses.
    {
      const __bf16* vsrc = Vh + base + (size_t)(k0 + lane) * Dc;
      const unsigned ldsa = (unsigned)(uintptr_t)&Vs[wid][lane][0];
#pragma unroll
      for (int i = 0; i < 4; ++i) {
        asm volatile("global_load_async_to_lds_b128 %0, %1, off offset:%2"
                     :: "v"(ldsa), "v"(vsrc), "i"(i * 16) : "memory");
      }
#pragma unroll
      for (int i = 4; i < 8; ++i) {
        asm volatile("global_load_async_to_lds_b128 %0, %1, off offset:%2"
                     :: "v"(ldsa), "v"(vsrc), "i"(i * 16) : "memory");
      }
    }

    // K^T B-fragments: lane holds key column (l16); per 32-d split, half h holds d 8h..8h+7 and 16+8h..16+8h+7.
    v16bf kbf[2][2];
#pragma unroll
    for (int nt = 0; nt < 2; ++nt) {
      const __bf16* kp = Kh + base + (size_t)(k0 + nt * 16 + l16) * Dc;
      kbf[nt][0] = frag_from_bf16(kp + half * 8,      kp + 16 + half * 8);
      kbf[nt][1] = frag_from_bf16(kp + 32 + half * 8, kp + 48 + half * 8);
    }

    v8f s[2][2];
#pragma unroll
    for (int mt = 0; mt < 2; ++mt)
#pragma unroll
      for (int nt = 0; nt < 2; ++nt) {
        v8f t = zero8();
        t = wmma_bf16(qa[mt][0], kbf[nt][0], t);
        t = wmma_bf16(qa[mt][1], kbf[nt][1], t);
        s[mt][nt] = t;
      }

    // Scale, causal mask, online softmax (rows live across 16-lane halves).
#pragma unroll
    for (int mt = 0; mt < 2; ++mt) {
#pragma unroll
      for (int r = 0; r < 8; ++r) {
        const int qi = qrow + mt * 16 + r + 8 * half;
        const float v0 = (k0 + l16      <= qi) ? s[mt][0][r] * 0.125f : NEG_INF;
        const float v1 = (k0 + 16 + l16 <= qi) ? s[mt][1][r] * 0.125f : NEG_INF;
        const float rm = rmax16(fmaxf(v0, v1));
        const float nm = fmaxf(m[mt][r], rm);
        alpha[mt][r] = __expf(m[mt][r] - nm);
        const float p0 = __expf(v0 - nm);
        const float p1 = __expf(v1 - nm);
        m[mt][r] = nm;
        l[mt][r] = l[mt][r] * alpha[mt][r] + rsum16(p0 + p1);
        Ps[wid][mt * 16 + r + 8 * half][l16]      = (__bf16)p0;  // D->LDS
        Ps[wid][mt * 16 + r + 8 * half][16 + l16] = (__bf16)p1;
      }
    }
    __threadfence_block();

    // Re-read P as A-layout fragments.
    v16bf pa[2];
#pragma unroll
    for (int mt = 0; mt < 2; ++mt)
      pa[mt] = frag_from_bf16(&Ps[wid][mt * 16 + l16][half * 16],
                              &Ps[wid][mt * 16 + l16][half * 16 + 8]);

    // Wait for the async V tile before building B fragments from LDS.
    asm volatile("s_wait_asynccnt 0x0" ::: "memory");

    // O = diag(alpha) * O + P @ V  (four 16-col output tiles, shared vb).
#pragma unroll
    for (int dt = 0; dt < 4; ++dt) {
      v16bf vb;  // B layout: K (= local key) strided over LDS rows
#pragma unroll
      for (int e = 0; e < 16; ++e)
        vb[e] = Vs[wid][half * 16 + e][dt * 16 + l16];
#pragma unroll
      for (int mt = 0; mt < 2; ++mt) {
#pragma unroll
        for (int r = 0; r < 8; ++r) O[mt][dt][r] *= alpha[mt][r];
        O[mt][dt] = wmma_bf16(pa[mt], vb, O[mt][dt]);
      }
    }
    __threadfence_block();
  }

  // Normalize and write Y (bf16, [B*T, C]).
#pragma unroll
  for (int mt = 0; mt < 2; ++mt) {
#pragma unroll
    for (int r = 0; r < 8; ++r) l[mt][r] = 1.0f / l[mt][r];
#pragma unroll
    for (int dt = 0; dt < 4; ++dt) {
#pragma unroll
      for (int r = 0; r < 8; ++r) {
        const int q = qrow + mt * 16 + r + 8 * half;
        Yb[(size_t)(b * Tc + q) * Cc + h * Dc + dt * 16 + l16] =
            (__bf16)(O[mt][dt][r] * l[mt][r]);
      }
    }
  }
}

// ---------------------------------------------------------------------------
// Kernel 4: out = Y @ W_proj^T + b_proj  (bf16 A, f32 weights, f32 out)
// Register-blocked 64x64 per wave like kernel 1.
// ---------------------------------------------------------------------------
__global__ __launch_bounds__(128) void proj_gemm(
    const __bf16* __restrict__ Yb, const float* __restrict__ Wp,
    const float* __restrict__ bp, float* __restrict__ out) {
  const int lane = threadIdx.x & 31, wid = threadIdx.x >> 5;
  const int half = lane >> 4, l16 = lane & 15;
  const int tM64 = blockIdx.x / (Cc / 256);
  const int nW   = (blockIdx.x % (Cc / 256)) * 256 + wid * 64;
  const int rowBase = tM64 * 64;

  const __bf16* yr[4];
  const float*  wr[4];
#pragma unroll
  for (int mt = 0; mt < 4; ++mt)
    yr[mt] = Yb + (size_t)(rowBase + mt * 16 + l16) * Cc;
#pragma unroll
  for (int nt = 0; nt < 4; ++nt)
    wr[nt] = Wp + (size_t)(nW + nt * 16 + l16) * Cc;

  v8f acc[4][4];
#pragma unroll
  for (int mt = 0; mt < 4; ++mt)
#pragma unroll
    for (int nt = 0; nt < 4; ++nt) acc[mt][nt] = zero8();

  for (int kb = 0; kb < Cc; kb += 32) {
    v16bf a[4], b[4];
#pragma unroll
    for (int mt = 0; mt < 4; ++mt)
      a[mt] = frag_from_bf16(yr[mt] + kb + half * 8,
                             yr[mt] + kb + 16 + half * 8);
#pragma unroll
    for (int nt = 0; nt < 4; ++nt)
      b[nt] = frag_from_f32(wr[nt] + kb + half * 8,
                            wr[nt] + kb + 16 + half * 8);
#pragma unroll
    for (int mt = 0; mt < 4; ++mt)
#pragma unroll
      for (int nt = 0; nt < 4; ++nt)
        acc[mt][nt] = wmma_bf16(a[mt], b[nt], acc[mt][nt]);
  }

#pragma unroll
  for (int nt = 0; nt < 4; ++nt) {
    const int n = nW + nt * 16 + l16;
    const float bias = bp[n];
#pragma unroll
    for (int mt = 0; mt < 4; ++mt) {
#pragma unroll
      for (int r = 0; r < 8; ++r) {
        const int row = rowBase + mt * 16 + r + 8 * half;
        out[(size_t)row * Cc + n] = acc[mt][nt][r] + bias;
      }
    }
  }
}

// ---------------------------------------------------------------------------
// Host launcher
// ---------------------------------------------------------------------------
extern "C" void kernel_launch(void* const* d_in, const int* in_sizes, int n_in,
                              void* d_out, int out_size, void* d_ws,
                              size_t ws_size, hipStream_t stream) {
  const float* x  = (const float*)d_in[0];
  const float* Wa = (const float*)d_in[1];
  const float* ba = (const float*)d_in[2];
  const float* Wp = (const float*)d_in[3];
  const float* bp = (const float*)d_in[4];
  float* out = (float*)d_out;

  // Workspace layout (bf16 elements):
  //   QKV [4096 x 3072] | Qh | Kh | Vh (each 4096x1024-equiv) | Y [4096 x 1024]
  __bf16* QKV = (__bf16*)d_ws;
  __bf16* Qh  = QKV + (size_t)Mrows * Nqkv;
  __bf16* Kh  = Qh + (size_t)Mrows * Cc;
  __bf16* Vh  = Kh + (size_t)Mrows * Cc;
  __bf16* Yb  = Vh + (size_t)Mrows * Cc;

  qkv_gemm<<<(Mrows / 64) * (Nqkv / 256), 128, 0, stream>>>(x, Wa, ba, QKV);
  rope_reshape<<<Bc * Hc * Tc, 64, 0, stream>>>(QKV, Qh, Kh, Vh);
  flash_attn<<<Bc * Hc * (Tc / 128), 128, 0, stream>>>(Qh, Kh, Vh, Yb);
  proj_gemm<<<(Mrows / 64) * (Cc / 256), 128, 0, stream>>>(Yb, Wp, bp, out);
}
